// MyGRU_12481174962906
// MI455X (gfx1250) — matrix operand order repaired and checked by reading, not compiled, not run on hardware
//
#include <hip/hip_runtime.h>
#include <stdint.h>

// ---------------------------------------------------------------------------
// Sequential GRU scan (2 layers x 2048 steps), persistent-grid, bf16 WMMA.
//   B=2,S=1024,I=1024,H=1024,L=2  ->  x_flat[2048][1024], one shared hidden.
// Weights converted once to bf16, packed into the exact WMMA B-fragment lane
// layout; they stay resident in the 192MB L2 and are streamed every step.
// Mat-vec on WMMA: input vector replicated across A's 16 rows, W^T tile in B.
// ---------------------------------------------------------------------------

typedef __attribute__((ext_vector_type(16))) __bf16 v16bf;
typedef __attribute__((ext_vector_type(8)))  float  v8f;

#define HDIM   1024
#define KDIM   2048           // I + H
#define LAYERS 2
#define STEPS  2048           // B*S, sequential
#define TILES  64             // HDIM / 16 output tiles
#define KCH    64             // KDIM / 32 k-fragments
#define NWG    64
#define NTHR   256            // 8 waves (wave32)
#define MAT_DWORDS (TILES * KCH * 32 * 8)   // 1,048,576 dwords = 4 MB / matrix

// workspace layout (bytes)
static constexpr size_t OFF_WPK = 0;                       // 6 matrices * 4MB = 24MB
static constexpr size_t OFF_EMB = (size_t)24 << 20;        // 2048*1024 bf16 = 4MB
static constexpr size_t OFF_HF  = (size_t)28 << 20;        // h (f32)   4KB
static constexpr size_t OFF_HB  = OFF_HF + 4096;           // h (bf16)
static constexpr size_t OFF_Z   = OFF_HB + 4096;           // z (f32)
static constexpr size_t OFF_RH  = OFF_Z  + 4096;           // r*h (bf16)
static constexpr size_t OFF_CTR = OFF_RH + 4096;           // grid-sync counter

// --- prep: f32 -> bf16 weight pack into WMMA B-fragment layout --------------
// B tile is 32x16 (k x n), packed in contiguous k order: lanes 0-15 hold K=0..15 (VGPR j -> K=2j,2j+1),
// lanes 16-31 hold K=16..31; column n = lane & 15.
__global__ void pack_weights(const float* __restrict__ Wr,
                             const float* __restrict__ Wz,
                             const float* __restrict__ Wh,
                             uint32_t*    __restrict__ wpk) {
    size_t idx = (size_t)blockIdx.x * blockDim.x + threadIdx.x;
    if (idx >= (size_t)6 * MAT_DWORDS) return;
    int    mat  = (int)(idx / MAT_DWORDS);
    size_t rem  = idx % MAT_DWORDS;
    int l    = mat / 3;
    int gate = mat % 3;                     // 0=r, 1=z, 2=h
    int tile = (int)(rem / (KCH * 32 * 8));
    int r2   = (int)(rem % (KCH * 32 * 8));
    int kc   = r2 / (32 * 8);
    int r3   = r2 % (32 * 8);
    int lane = r3 / 8;
    int j    = r3 % 8;
    int n  = tile * 16 + (lane & 15);
    int k0 = kc * 32 + (lane >> 4) * 16 + 2 * j;
    const float* W = (gate == 0) ? Wr : ((gate == 1) ? Wz : Wh);
    const float* s = W + ((size_t)l * HDIM + n) * KDIM + k0;
    unsigned short u0 = __builtin_bit_cast(unsigned short, (__bf16)s[0]);
    unsigned short u1 = __builtin_bit_cast(unsigned short, (__bf16)s[1]);
    wpk[idx] = (uint32_t)u0 | ((uint32_t)u1 << 16);
}

__global__ void conv_emb(const float* __restrict__ e, __bf16* __restrict__ o, int n) {
    int i = blockIdx.x * blockDim.x + threadIdx.x;
    if (i < n) o[i] = (__bf16)e[i];
}

__global__ void gru_init(float* hf, __bf16* hb, float* zf, __bf16* rh, uint32_t* ctr) {
    int i = blockIdx.x * blockDim.x + threadIdx.x;
    if (i < HDIM) { hf[i] = 0.f; hb[i] = (__bf16)0.f; zf[i] = 0.f; rh[i] = (__bf16)0.f; }
    if (i == 0) *ctr = 0u;
}

// --- fast nonlinearities on the v_exp/v_rcp transcendental units ------------
__device__ __forceinline__ float fast_sigmoid(float x) {
    return __builtin_amdgcn_rcpf(1.f + __expf(-x));
}
__device__ __forceinline__ float fast_tanh(float x) {
    // limits: x->+inf: exp->inf, rcp->0 => 1 ; x->-inf: exp->0 => -1
    return 1.f - 2.f * __builtin_amdgcn_rcpf(__expf(2.f * x) + 1.f);
}

// --- grid-wide split barrier via agent-scope atomic ------------------------
__device__ __forceinline__ void grid_sync(uint32_t* ctr, uint32_t& target) {
    __threadfence();            // make this thread's global stores visible (agent)
    __syncthreads();
    target += NWG;
    if (threadIdx.x == 0) {
        __hip_atomic_fetch_add(ctr, 1u, __ATOMIC_ACQ_REL, __HIP_MEMORY_SCOPE_AGENT);
        while (__hip_atomic_load(ctr, __ATOMIC_ACQUIRE, __HIP_MEMORY_SCOPE_AGENT) < target)
            __builtin_amdgcn_s_sleep(2);
    }
    __syncthreads();
}

// --- A fragment: input vector replicated across all 16 rows ----------------
// A 16x32 in this code's contiguous k order: VGPR j, lanes<16: K=2j ; lanes>=16: +16 (matches B pack)
// (compiles to two contiguous ds_load_b128 per fragment)
__device__ __forceinline__ v16bf make_a(const __bf16* vec, int kb, int lhi) {
    v16bf a;
#pragma unroll
    for (int j = 0; j < 8; ++j) {
        int koff = (lhi << 4) + (j << 1);
        a[2 * j]     = vec[kb + koff];
        a[2 * j + 1] = vec[kb + koff + 1];
    }
    return a;
}

__device__ __forceinline__ v16bf load_b(const uint32_t* p) {
    union { uint4 u[2]; v16bf v; } B;
    B.u[0] = ((const uint4*)p)[0];
    B.u[1] = ((const uint4*)p)[1];
    return B.v;
}

// --- persistent GRU scan ----------------------------------------------------
__global__ __launch_bounds__(NTHR) void gru_persistent(
    const uint32_t* __restrict__ wpk, const __bf16* __restrict__ emb,
    const float* __restrict__ br, const float* __restrict__ bz,
    const float* __restrict__ bh,
    float* __restrict__ hf, __bf16* __restrict__ hb,
    float* __restrict__ zf, __bf16* __restrict__ rh,
    uint32_t* __restrict__ ctr, float* __restrict__ out) {

    __shared__ __bf16 vec[KDIM];      // [h | x] or [r*h | x], 4KB
    __shared__ float  part[8 * 16];   // per-wave partial outputs

    const int tid  = threadIdx.x;
    const int g    = blockIdx.x;      // output tile (16 rows) owned by this WG
    const int wave = tid >> 5;
    const int lane = tid & 31;
    const int l15  = lane & 15;
    const int lhi  = lane >> 4;
    uint32_t target = 0;

    for (int l = 0; l < LAYERS; ++l) {
        const float* brl = br + l * HDIM;
        const float* bzl = bz + l * HDIM;
        const float* bhl = bh + l * HDIM;
        // phase-1: waves 0-3 -> r gate, waves 4-7 -> z gate; 4-way K split
        const int gate = wave >> 2;
        const int ksub = wave & 3;
        const uint32_t* w1 = wpk + (size_t)(l * 3 + gate) * MAT_DWORDS
                                 + (size_t)g * (KCH * 256)
                                 + (size_t)(ksub * 16) * 256 + lane * 8;
        const uint32_t* w2 = wpk + (size_t)(l * 3 + 2) * MAT_DWORDS
                                 + (size_t)g * (KCH * 256)
                                 + (size_t)(wave * 8) * 256 + lane * 8;

        for (int t = 0; t < STEPS; ++t) {
            const __bf16* x = emb + (size_t)t * HDIM;

            // ---------------- phase 1: r,z = sigmoid(W @ [h,x] + b) --------
            for (int i = tid; i < HDIM; i += NTHR) { vec[i] = hb[i]; vec[HDIM + i] = x[i]; }
            if (t + 1 < STEPS) __builtin_prefetch(x + HDIM + tid * 4, 0, 1);
            __syncthreads();

            {
                v8f acc = {0.f, 0.f, 0.f, 0.f, 0.f, 0.f, 0.f, 0.f};
                const uint32_t* wp = w1;
#pragma unroll 4
                for (int f = 0; f < 16; ++f) {
                    v16bf a = make_a(vec, ksub * 512 + f * 32, lhi);
                    v16bf b = load_b(wp);
                    acc = __builtin_amdgcn_wmma_f32_16x16x32_bf16(
                        false, a, false, b, (short)0, acc, false, false);
                    wp += 256;
                }
                if (lane < 16) part[wave * 16 + l15] = acc[0];
            }
            __syncthreads();
            if (tid < 32) {
                int gt = tid >> 4, i = tid & 15, idx = g * 16 + i;
                float s = part[(gt * 4 + 0) * 16 + i] + part[(gt * 4 + 1) * 16 + i]
                        + part[(gt * 4 + 2) * 16 + i] + part[(gt * 4 + 3) * 16 + i];
                s += (gt == 0) ? brl[idx] : bzl[idx];
                float sig = fast_sigmoid(s);
                if (gt == 0) rh[idx] = (__bf16)(sig * hf[idx]);   // r*h
                else         zf[idx] = sig;                        // z
            }
            grid_sync(ctr, target);

            // ---------------- phase 2: hh = tanh(Wh @ [r*h, x] + bh) -------
            for (int i = tid; i < HDIM; i += NTHR) { vec[i] = rh[i]; vec[HDIM + i] = x[i]; }
            __syncthreads();

            {
                v8f acc = {0.f, 0.f, 0.f, 0.f, 0.f, 0.f, 0.f, 0.f};
                const uint32_t* wp = w2;
#pragma unroll 4
                for (int f = 0; f < 8; ++f) {
                    v16bf a = make_a(vec, wave * 256 + f * 32, lhi);
                    v16bf b = load_b(wp);
                    acc = __builtin_amdgcn_wmma_f32_16x16x32_bf16(
                        false, a, false, b, (short)0, acc, false, false);
                    wp += 256;
                }
                if (lane < 16) part[wave * 16 + l15] = acc[0];
            }
            __syncthreads();
            if (tid < 16) {
                int idx = g * 16 + tid;
                float s = bhl[idx];
#pragma unroll
                for (int q = 0; q < 8; ++q) s += part[q * 16 + tid];
                float hhv = fast_tanh(s);
                float zz  = zf[idx];
                float hn  = (1.f - zz) * hf[idx] + zz * hhv;
                hf[idx] = hn;
                hb[idx] = (__bf16)hn;
            }
            grid_sync(ctr, target);
        }
    }

    if (g == 0)
        for (int i = tid; i < HDIM; i += NTHR) { out[i] = hf[i]; out[HDIM + i] = hf[i]; }
}

// ---------------------------------------------------------------------------
extern "C" void kernel_launch(void* const* d_in, const int* in_sizes, int n_in,
                              void* d_out, int out_size, void* d_ws, size_t ws_size,
                              hipStream_t stream) {
    (void)in_sizes; (void)n_in; (void)out_size; (void)ws_size;
    const float* emb = (const float*)d_in[0];
    const float* Wr  = (const float*)d_in[1];
    const float* br  = (const float*)d_in[2];
    const float* Wz  = (const float*)d_in[3];
    const float* bz  = (const float*)d_in[4];
    const float* Wh  = (const float*)d_in[5];
    const float* bh  = (const float*)d_in[6];
    float* out = (float*)d_out;

    uint8_t*  ws   = (uint8_t*)d_ws;
    uint32_t* wpk  = (uint32_t*)(ws + OFF_WPK);
    __bf16*   embb = (__bf16*)  (ws + OFF_EMB);
    float*    hf   = (float*)   (ws + OFF_HF);
    __bf16*   hb   = (__bf16*)  (ws + OFF_HB);
    float*    zf   = (float*)   (ws + OFF_Z);
    __bf16*   rhv  = (__bf16*)  (ws + OFF_RH);
    uint32_t* ctr  = (uint32_t*)(ws + OFF_CTR);

    const int nEmb = STEPS * HDIM;                       // 2M elements
    pack_weights<<<(6 * MAT_DWORDS + 255) / 256, 256, 0, stream>>>(Wr, Wz, Wh, wpk);
    conv_emb<<<(nEmb + 255) / 256, 256, 0, stream>>>(emb, embb, nEmb);
    gru_init<<<(HDIM + 255) / 256, 256, 0, stream>>>(hf, hb, zf, rhv, ctr);
    gru_persistent<<<NWG, NTHR, 0, stream>>>(wpk, embb, br, bz, bh,
                                             hf, hb, zf, rhv, ctr, out);
}
